// PGExplainer_28123445854360
// MI455X (gfx1250) — matrix operand order repaired and checked by reading, not compiled, not run on hardware
//
#include <hip/hip_runtime.h>
#include <hip/hip_bf16.h>

typedef __attribute__((ext_vector_type(16))) _Float16 v16h;
typedef __attribute__((ext_vector_type(8)))  float    v8f;

#define NODES  50000
#define FDIM   256
#define EDGES  800000
#define LAYERS 3
#define HDIM   64
#define PCOLS  (LAYERS * 2 * HDIM)   // 384 columns of P: [l*128 + s*64 + h]
#define KTILES (FDIM / 32)           // 8 k-steps of 32
#define NTILES (PCOLS / 16)          // 24
#define MTILES (NODES / 16)          // 3125 (exact: 50000 = 16*3125)
#define NTPW   4                     // N-tiles per wave (A-fragment reuse factor)
#define NGROUPS (NTILES / NTPW)      // 6
#define TOTAL_WAVES (MTILES * NGROUPS)  // 18750
#define WPACK_ELTS (NTILES * KTILES * 512)

// ---------------------------------------------------------------------------
// Pack W1 [L, 2F, H] (f32) into a contiguous-k v_wmma_f32_16x16x32_f16 B-matrix
// lane layout, one 512-half block per (ntile, ktile):
//   lane l holds B[k = kt*32 + (l>>4)*16 + i][n = l&15], i = 0..15 contiguous.
// ---------------------------------------------------------------------------
__global__ void __launch_bounds__(256) pack_w1(const float* __restrict__ W1,
                                               _Float16* __restrict__ wpack) {
    int idx = blockIdx.x * 256 + threadIdx.x;
    if (idx >= WPACK_ELTS) return;
    int nt  = idx / (KTILES * 512);
    int rem = idx % (KTILES * 512);
    int kt  = rem / 512;
    int q   = rem % 512;
    int l   = q / 16;                    // lane 0..31
    int i   = q % 16;                    // element within lane
    int n   = l & 15;                    // B column within tile
    int k   = kt * 32 + (l >> 4) * 16 + i;
    int c   = nt * 16 + n;               // global P column
    int lay = c >> 7;                    // layer
    int r2  = c & 127;
    int s   = r2 >> 6;                   // 0 = row-half (W1[:,:F]), 1 = col-half (W1[:,F:])
    int h   = r2 & 63;
    wpack[idx] = (_Float16)W1[(size_t)lay * (2 * FDIM * HDIM) +
                              (size_t)(s * FDIM + k) * HDIM + h];
}

// ---------------------------------------------------------------------------
// P[n, 0:384] = x[n, 0:256] @ Wcat  via v_wmma_f32_16x16x32_f16.
// One wave computes NTPW=4 adjacent 16x16 N-tiles of one M-row group,
// reusing a single A fragment (x rows, converted f32->f16 once) for 4 WMMAs
// per k-step: 32 WMMAs per wave, A loads/conversions amortized 4x.
// ---------------------------------------------------------------------------
__global__ void __launch_bounds__(256) node_proj_wmma(const float* __restrict__ x,
                                                      const _Float16* __restrict__ wpack,
                                                      float* __restrict__ P) {
    int gwave = (blockIdx.x * 256 + threadIdx.x) >> 5;
    int lane  = threadIdx.x & 31;
    if (gwave >= TOTAL_WAVES) return;            // wave-uniform: EXEC stays all-ones
    int mt    = gwave / NGROUPS;
    int nt0   = (gwave % NGROUPS) * NTPW;

    int khalf = lane >> 4;                       // which K-half this lane holds
    int row   = mt * 16 + (lane & 15);           // A-matrix row for this lane
    const float*    xrow = x + (size_t)row * FDIM + khalf * 16;
    const _Float16* wp   = wpack + (size_t)nt0 * KTILES * 512 + lane * 16;

    v8f acc[NTPW];
#pragma unroll
    for (int t = 0; t < NTPW; ++t) acc[t] = (v8f){};

#pragma unroll
    for (int kt = 0; kt < KTILES; ++kt) {
        // A fragment (contiguous k order, same as wpack B): lanes<16 hold K {0..15},
        // lanes>=16 hold K {16..31}; converted f32 -> f16 on the fly.
        const float4 a0 = *(const float4*)(xrow + kt * 32 + 0);
        const float4 a1 = *(const float4*)(xrow + kt * 32 + 4);
        const float4 a2 = *(const float4*)(xrow + kt * 32 + 8);
        const float4 a3 = *(const float4*)(xrow + kt * 32 + 12);
        v16h a;
        a[0]  = (_Float16)a0.x; a[1]  = (_Float16)a0.y;
        a[2]  = (_Float16)a0.z; a[3]  = (_Float16)a0.w;
        a[4]  = (_Float16)a1.x; a[5]  = (_Float16)a1.y;
        a[6]  = (_Float16)a1.z; a[7]  = (_Float16)a1.w;
        a[8]  = (_Float16)a2.x; a[9]  = (_Float16)a2.y;
        a[10] = (_Float16)a2.z; a[11] = (_Float16)a2.w;
        a[12] = (_Float16)a3.x; a[13] = (_Float16)a3.y;
        a[14] = (_Float16)a3.z; a[15] = (_Float16)a3.w;

#pragma unroll
        for (int t = 0; t < NTPW; ++t) {
            // wpack is tiny (192 KB) -> L0/L2 resident
            v16h b = *(const v16h*)(wp + ((size_t)t * KTILES + kt) * 512);
            acc[t] = __builtin_amdgcn_wmma_f32_16x16x32_f16(
                /*neg_a=*/false, a, /*neg_b=*/false, b,
                /*c_mod=*/(short)0, acc[t], /*reuse_a=*/false, /*reuse_b=*/false);
        }
    }

    // D layout: VGPR v, lane L -> row (v + 8*(L>>4)), col (L&15)
    int mb = mt * 16 + 8 * khalf;
#pragma unroll
    for (int t = 0; t < NTPW; ++t) {
        int nb = (nt0 + t) * 16 + (lane & 15);
#pragma unroll
        for (int v = 0; v < 8; ++v)
            P[(size_t)(mb + v) * PCOLS + nb] = acc[t][v];
    }
}

// ---------------------------------------------------------------------------
// Edge pass: gather P rows (L2-resident), fused bias+relu+dot+sigmoid.
// ---------------------------------------------------------------------------
__global__ void __launch_bounds__(256) edge_mlp(const int* __restrict__ ei,
                                                const float* __restrict__ P,
                                                const float* __restrict__ b1,
                                                const float* __restrict__ W2,
                                                const float* __restrict__ b2,
                                                float* __restrict__ out) {
    __shared__ __align__(16) float sb1[LAYERS * HDIM];
    __shared__ __align__(16) float sW2[LAYERS * HDIM];
    __shared__ float sb2[LAYERS];
    for (int t = threadIdx.x; t < LAYERS * HDIM; t += 256) {
        sb1[t] = b1[t];
        sW2[t] = W2[t];
    }
    if (threadIdx.x < LAYERS) sb2[threadIdx.x] = b2[threadIdx.x];
    __syncthreads();

    int e = blockIdx.x * 256 + threadIdx.x;
    if (e >= EDGES) return;
    int r = ei[e];
    int c = ei[EDGES + e];
    const float* pr = P + (size_t)r * PCOLS;          // row-half columns at l*128 + 0..63
    const float* pc = P + (size_t)c * PCOLS + HDIM;   // col-half columns at l*128 + 64..127

    float acc = 0.0f;
#pragma unroll
    for (int l = 0; l < LAYERS; ++l) {
        float dot = sb2[l];
        const float* prl = pr + l * 2 * HDIM;
        const float* pcl = pc + l * 2 * HDIM;
        const float* bl  = sb1 + l * HDIM;
        const float* wl  = sW2 + l * HDIM;
#pragma unroll
        for (int j = 0; j < HDIM; j += 4) {
            float4 av  = *(const float4*)(prl + j);
            float4 bv  = *(const float4*)(pcl + j);
            float4 biv = *(const float4*)(bl + j);
            float4 wv  = *(const float4*)(wl + j);
            float h;
            h = fmaxf(av.x + bv.x + biv.x, 0.0f); dot = fmaf(h, wv.x, dot);
            h = fmaxf(av.y + bv.y + biv.y, 0.0f); dot = fmaf(h, wv.y, dot);
            h = fmaxf(av.z + bv.z + biv.z, 0.0f); dot = fmaf(h, wv.z, dot);
            h = fmaxf(av.w + bv.w + biv.w, 0.0f); dot = fmaf(h, wv.w, dot);
        }
        acc += 1.0f / (1.0f + __expf(-dot));
    }
    out[e] = acc * (1.0f / 3.0f);
}

// ---------------------------------------------------------------------------
extern "C" void kernel_launch(void* const* d_in, const int* in_sizes, int n_in,
                              void* d_out, int out_size, void* d_ws, size_t ws_size,
                              hipStream_t stream) {
    const float* x  = (const float*)d_in[0];   // [N, F]
    const int*   ei = (const int*)d_in[1];     // [2, E]
    const float* W1 = (const float*)d_in[2];   // [L, 2F, H]
    const float* b1 = (const float*)d_in[3];   // [L, H]
    const float* W2 = (const float*)d_in[4];   // [L, H, 1]
    const float* b2 = (const float*)d_in[5];   // [L, 1]
    float* out = (float*)d_out;

    char* ws = (char*)d_ws;
    _Float16* wpack = (_Float16*)ws;                               // 196,608 B
    size_t wpack_bytes = ((size_t)WPACK_ELTS * sizeof(_Float16) + 255) & ~(size_t)255;
    float* P = (float*)(ws + wpack_bytes);                         // 76.8 MB, L2-resident

    pack_w1<<<(WPACK_ELTS + 255) / 256, 256, 0, stream>>>(W1, wpack);
    node_proj_wmma<<<(TOTAL_WAVES + 7) / 8, 256, 0, stream>>>(x, wpack, P);  // 2344 blocks, 8 waves each
    edge_mlp<<<EDGES / 256, 256, 0, stream>>>(ei, P, b1, W2, b2, out);       // 3125 blocks
}
